// LGP_30571577213079
// MI455X (gfx1250) — compile-verified, matrix operand order checked by reading, not run on hardware
//
#include <hip/hip_runtime.h>

#define N_PTS 16384
#define DIMF  192
#define NSW   16
#define NHEAD 6
#define HD    32

// workspace layout (float offsets)
#define OFF_SUM 0
#define OFF_MAX 1
#define OFF_S1  2
#define OFF_S2  64
#define OFF_VOL 2368
#define OFF_BNS 2560
#define OFF_BNF 3136
#define OFF_PRE 4096
#define OFF_XF  (OFF_PRE + N_PTS*DIMF)
#define OFF_AO  (OFF_XF  + N_PTS*DIMF)
#define OFF_WT  (OFF_AO  + N_PTS*DIMF)   // multiple of 8 floats -> 32B aligned

// bf16-transposed weight region (ushort offsets from wt base); layout [batch][col][ldk]
#define WT_QKV   0                       // [576][192]
#define WT_PROJ  (WT_QKV  + 576*192)     // [192][192]
#define WT_PROP  (WT_PROJ + 192*192)     // [192][224]  K padded 195->224
#define WT_DE2   (WT_PROP + 192*224)     // [192][64]   K padded 48->64
#define WT_FUS   (WT_DE2  + 192*64)      // [192][192]
#define WT_PW1   (WT_FUS  + 192*192)     // [3][192][64]
#define WT_PW2   (WT_PW1  + 3*192*64)    // [3][192][192]

typedef __attribute__((ext_vector_type(16))) __bf16         v16bf;
typedef __attribute__((ext_vector_type(16))) unsigned short v16us;
typedef __attribute__((ext_vector_type(8)))  float          v8f;

__device__ __forceinline__ unsigned short f2bfu(float f) {
    union { __bf16 b; unsigned short u; } c;
    c.b = (__bf16)f;                       // hardware v_cvt
    return c.u;
}
__device__ __forceinline__ float bfu2f(unsigned short s) {
    return __uint_as_float(((unsigned)s) << 16);
}

// A fragment: 16xK tile from LDS bf16; lane&15 = M row, lane>=16 -> K+16. 32B vector load.
__device__ __forceinline__ v16bf load_a_frag(const unsigned short* s, int ld, int k0, int lane) {
    union { v16us u; v16bf b; } c;
    c.u = *(const v16us*)(s + (lane & 15) * ld + k0 + ((lane >> 4) << 4));
    return c.b;
}
// B fragment from pre-transposed bf16 weights Wt[col][ldk]; lane&15 = N column. 32B vector load.
__device__ __forceinline__ v16bf load_bt_frag(const unsigned short* Wt, int ldk, int k0, int n0,
                                              int lane) {
    union { v16us u; v16bf b; } c;
    c.u = *(const v16us*)(Wt + (n0 + (lane & 15)) * ldk + k0 + ((lane >> 4) << 4));
    return c.b;
}
__device__ __forceinline__ v8f wmma_bf16(v16bf a, v16bf b, v8f c) {
    return __builtin_amdgcn_wmma_f32_16x16x32_bf16(false, a, false, b, (short)0, c, false, false);
}

// ---------------- K0a: zero the stats region ----------------
__global__ void LGP_init(float* ws) {
    int i = blockIdx.x * blockDim.x + threadIdx.x;
    if (i < OFF_PRE) ws[i] = 0.f;
}

// ---------------- K0b: transpose + convert weights to bf16 [b][col][ldk] ----------------
__global__ void LGP_prep(const float* __restrict__ src, unsigned short* __restrict__ dst,
                         int K, int Ncols, int ldk, int nbatch) {
    int total = nbatch * Ncols * ldk;
    for (int i = blockIdx.x * blockDim.x + threadIdx.x; i < total; i += gridDim.x * blockDim.x) {
        int b = i / (Ncols * ldk);
        int rem = i % (Ncols * ldk);
        int col = rem / ldk, k = rem % ldk;
        float v = (k < K) ? src[b * K * Ncols + k * Ncols + col] : 0.f;
        dst[i] = f2bfu(v);
    }
}

// ---------------- K1: per-window prefeatures ----------------
__global__ void __launch_bounds__(64) LGP_k1(
    const float* __restrict__ p, const float* __restrict__ x, const int* __restrict__ knn,
    const float* __restrict__ prop_b,
    const float* __restrict__ gA_W, const float* __restrict__ gA_b,
    const float* __restrict__ gB_W, const float* __restrict__ gB_b,
    const float* __restrict__ kn,
    const float* __restrict__ de_W1, const float* __restrict__ de_b1,
    const float* __restrict__ de_b2,
    const float* __restrict__ kp_W, const float* __restrict__ kp_b,
    float* ws) {
    __shared__ float bS1[48];
    __shared__ float bS2[48 * 48];
    __shared__ unsigned short __align__(32) Aprop[2][16 * 224];
    __shared__ unsigned short __align__(32) Adf[2][16 * 64];
    __shared__ float dfl[2][16 * 192];
    __shared__ float msum[2][192];
    __shared__ float kxyz[2][16 * 3];
    __shared__ float prl[2][16 * 3];
    __shared__ float dl[2][16];
    __shared__ float kdl[2][16];
    __shared__ float colred[2][32];
    __shared__ float smv[2][16];
    __shared__ int   widx[2][16];

    int lane = threadIdx.x & 31, w = threadIdx.x >> 5, tid = threadIdx.x;
    int n = blockIdx.x * 2 + w;
    const unsigned short* wt = (const unsigned short*)(ws + OFF_WT);
    for (int e = tid; e < 48 * 48; e += 64) bS2[e] = 0.f;
    for (int e = tid; e < 48; e += 64) bS1[e] = 0.f;
    __syncthreads();

    float pn0 = p[n * 3], pn1 = p[n * 3 + 1], pn2 = p[n * 3 + 2];
    if (lane < 16) widx[w][lane] = knn[n * NSW + lane];
    __syncthreads();
    if (lane < 16) {
        int j = widx[w][lane];
        float a0 = p[j * 3], a1 = p[j * 3 + 1], a2 = p[j * 3 + 2];
        kxyz[w][lane * 3] = a0; kxyz[w][lane * 3 + 1] = a1; kxyz[w][lane * 3 + 2] = a2;
        float r0 = a0 - pn0, r1 = a1 - pn1, r2 = a2 - pn2;
        prl[w][lane * 3] = r0; prl[w][lane * 3 + 1] = r1; prl[w][lane * 3 + 2] = r2;
        dl[w][lane] = sqrtf(r0 * r0 + r1 * r1 + r2 * r2 + 1e-12f);
    }
    __syncthreads();
    if (lane == 0) {
        float s = 0.f, mx = 0.f;
        for (int j = 0; j < 16; j++) { s += dl[w][j]; mx = fmaxf(mx, dl[w][j]); }
        atomicAdd(&ws[OFF_SUM], s);
        atomicMax((unsigned int*)&ws[OFF_MAX], __float_as_uint(mx));
    }
    // BN stats S1/S2 of pairwise rel-coord rows -> shared atomics
    for (int e = lane; e < 48 * 48; e += 32) {
        int a = e / 48, b = e % 48;
        int ja = a / 3, ca = a % 3, jb = b / 3, cb = b % 3;
        float va = kxyz[w][ja * 3 + ca], vb = kxyz[w][jb * 3 + cb];
        float acc = 0.f;
        for (int s = 0; s < 16; s++)
            acc += (va - kxyz[w][s * 3 + ca]) * (vb - kxyz[w][s * 3 + cb]);
        atomicAdd(&bS2[e], acc);
    }
    for (int e = lane; e < 48; e += 32) {
        int j = e / 3, c = e % 3;
        float vj = kxyz[w][j * 3 + c], acc = 0.f;
        for (int s = 0; s < 16; s++) acc += vj - kxyz[w][s * 3 + c];
        atomicAdd(&bS1[e], acc);
    }
    // stage A = [x_knn | p_r | 0-pad] (16 x 224)
    for (int t = lane; t < 16 * 224; t += 32) {
        int r = t / 224, c = t % 224;
        float v = 0.f;
        if (c < 192)      v = x[widx[w][r] * DIMF + c];
        else if (c < 195) v = prl[w][r * 3 + (c - 192)];
        Aprop[w][t] = f2bfu(v);
    }
    for (int c = lane; c < 192; c += 32) msum[w][c] = 0.f;
    __syncthreads();
    // propagate GEMM + bias + relu + mean over neighbors
    for (int n0 = 0; n0 < 192; n0 += 16) {
        v8f cc = {};
        for (int k0 = 0; k0 < 224; k0 += 32) {
            v16bf a = load_a_frag(Aprop[w], 224, k0, lane);
            v16bf b = load_bt_frag(wt + WT_PROP, 224, k0, n0, lane);
            cc = wmma_bf16(a, b, cc);
        }
        int col = n0 + (lane & 15);
        float bias = prop_b[col];
        float ps = 0.f;
#pragma unroll
        for (int r = 0; r < 8; r++) { float v = cc[r] + bias; ps += (v > 0.f ? v : 0.f); }
        colred[w][lane] = ps;
        __syncthreads();
        if (lane < 16) msum[w][n0 + lane] = (colred[w][lane] + colred[w][lane + 16]) * (1.f / 16.f);
        __syncthreads();
    }
    // keypoint distance features
    if (lane < 16) {
        float d0 = pn0 - kn[lane * 3], d1 = pn1 - kn[lane * 3 + 1], d2 = pn2 - kn[lane * 3 + 2];
        kdl[w][lane] = sqrtf(d0 * d0 + d1 * d1 + d2 * d2 + 1e-12f);
    }
    __syncthreads();
    for (int t = lane; t < 16 * 64; t += 32) {
        int j = t / 64, l = t % 64;
        float v = 0.f;
        if (l < 48) { v = kdl[w][j] * de_W1[l] + de_b1[l]; v = v > 0.f ? v : 0.f; }
        Adf[w][t] = f2bfu(v);
    }
    __syncthreads();
    for (int n0 = 0; n0 < 192; n0 += 16) {
        v8f cc = {};
        for (int k0 = 0; k0 < 64; k0 += 32) {
            v16bf a = load_a_frag(Adf[w], 64, k0, lane);
            v16bf b = load_bt_frag(wt + WT_DE2, 64, k0, n0, lane);
            cc = wmma_bf16(a, b, cc);
        }
        int col = n0 + (lane & 15), rb = (lane >> 4) * 8;
        float bias = de_b2[col];
#pragma unroll
        for (int r = 0; r < 8; r++) dfl[w][(rb + r) * 192 + col] = cc[r] + bias;
    }
    __syncthreads();
    // keypoint attention pooling (softmax over KSP)
    float rawsc = 0.f;
    if (lane < 16) {
        float s = kp_b[0];
        for (int c = 0; c < 192; c++) s += dfl[w][lane * 192 + c] * kp_W[c];
        rawsc = s;
    }
    __syncthreads();
    if (lane < 16) smv[w][lane] = rawsc;
    __syncthreads();
    float aval = 0.f;
    if (lane < 16) {
        float mx = -1e30f;
        for (int j = 0; j < 16; j++) mx = fmaxf(mx, smv[w][j]);
        float den = 0.f;
        for (int j = 0; j < 16; j++) den += expf(smv[w][j] - mx);
        aval = expf(smv[w][lane] - mx) / den;
    }
    __syncthreads();
    if (lane < 16) smv[w][lane] = aval;
    __syncthreads();
    // geo features + combined pre-fusion vector
    float m0 = 0, m1 = 0, m2 = 0, k0m = 0, k1m = 0, k2m = 0, gd = 0;
    for (int s = 0; s < 16; s++) {
        m0 += prl[w][s * 3]; m1 += prl[w][s * 3 + 1]; m2 += prl[w][s * 3 + 2];
        k0m += kxyz[w][s * 3]; k1m += kxyz[w][s * 3 + 1]; k2m += kxyz[w][s * 3 + 2];
        gd = fmaxf(gd, dl[w][s]);
    }
    float geo9[9] = {m0 / 16.f, m1 / 16.f, m2 / 16.f, k0m / 16.f, k1m / 16.f, k2m / 16.f,
                     pn0, pn1, pn2};
    float* pre = ws + OFF_PRE;
    for (int c = lane; c < 192; c += 32) {
        float dfeat = 0.f;
        for (int j = 0; j < 16; j++) dfeat += smv[w][j] * dfl[w][j * 192 + c];
        float g = gA_b[c];
        for (int a = 0; a < 9; a++) g += geo9[a] * gA_W[a * 192 + c];
        pre[n * DIMF + c] = x[n * DIMF + c] + msum[w][c] + g + gd * gB_W[c] + gB_b[c] + dfeat;
    }
    __syncthreads();
    for (int e = tid; e < 48 * 48; e += 64) atomicAdd(&ws[OFF_S2 + e], bS2[e]);
    for (int e = tid; e < 48; e += 64) atomicAdd(&ws[OFF_S1 + e], bS1[e]);
}

// ---------------- K2: finalize vol vector + BatchNorm params ----------------
__global__ void LGP_k2(const float* __restrict__ gC_W, const float* __restrict__ gC_b,
                       const float* __restrict__ pos_W1, const float* __restrict__ pos_b1,
                       const float* __restrict__ pos_gamma, const float* __restrict__ pos_beta,
                       float* ws) {
    int tid = threadIdx.x;
    const float Mtot = (float)(N_PTS * NSW);
    float mean = ws[OFF_SUM] / Mtot;
    float mx = __uint_as_float(*(const unsigned int*)&ws[OFF_MAX]);
    float vol = mean / (mx + 1e-8f);
    for (int c = tid; c < 192; c += 256) ws[OFF_VOL + c] = vol * gC_W[c] + gC_b[c];
    const float* S1 = &ws[OFF_S1];
    const float* S2 = &ws[OFF_S2];
    for (int ch = tid; ch < 3 * 192; ch += 256) {
        int i = ch / 192, c = ch % 192;
        const float* wv = &pos_W1[i * 48 * 192 + c];   // stride 192 down K
        float s1d = 0.f;
        for (int l = 0; l < 48; l++) s1d += wv[l * 192] * S1[l];
        float q = 0.f;
        for (int a = 0; a < 48; a++) {
            float wa = wv[a * 192], t = 0.f;
            for (int b = 0; b < 48; b++) t += S2[a * 48 + b] * wv[b * 192];
            q += wa * t;
        }
        float b1 = pos_b1[ch];
        float sum_h  = s1d + Mtot * b1;
        float sum_h2 = q + 2.f * b1 * s1d + Mtot * b1 * b1;
        float mu  = sum_h / Mtot;
        float var = sum_h2 / Mtot - mu * mu;
        float scl = pos_gamma[ch] * rsqrtf(var + 1e-5f);
        ws[OFF_BNS + ch] = scl;
        ws[OFF_BNF + ch] = pos_beta[ch] - mu * scl;
    }
}

// ---------------- K3/K5: [rows,192] x [192,192] WMMA GEMM + bias ----------------
__global__ void __launch_bounds__(256) LGP_gemm192(
    const float* __restrict__ A, const float* __restrict__ addvec,
    const unsigned short* __restrict__ Wt, const float* __restrict__ bias,
    float* __restrict__ out) {
    __shared__ unsigned short __align__(32) At[8][16 * 192];
    int lane = threadIdx.x & 31, w = threadIdx.x >> 5;
    int row0 = (blockIdx.x * 8 + w) * 16;
    for (int t = lane; t < 16 * 192; t += 32) {
        int r = t / 192, c = t % 192;
        float v = A[(row0 + r) * 192 + c];
        if (addvec) v += addvec[c];
        At[w][t] = f2bfu(v);
    }
    __syncthreads();
    for (int n0 = 0; n0 < 192; n0 += 16) {
        v8f cc = {};
        for (int k0 = 0; k0 < 192; k0 += 32) {
            v16bf a = load_a_frag(At[w], 192, k0, lane);
            v16bf b = load_bt_frag(Wt, 192, k0, n0, lane);
            cc = wmma_bf16(a, b, cc);
        }
        int col = n0 + (lane & 15), rb = (lane >> 4) * 8;
        float bv = bias[col];
#pragma unroll
        for (int r = 0; r < 8; r++) out[(row0 + rb + r) * 192 + col] = cc[r] + bv;
    }
}

// ---------------- K4: pos-MLPs + QKV + windowed attention ----------------
__global__ void __launch_bounds__(32) LGP_k4(
    const float* __restrict__ p, const int* __restrict__ knn,
    const float* __restrict__ pos_b1, const float* __restrict__ pos_b2,
    float* ws) {
    __shared__ unsigned short __align__(32) posb[16 * 64];
    __shared__ unsigned short __align__(32) hnb[16 * 192];
    __shared__ unsigned short __align__(32) wxb[16 * 192];
    __shared__ unsigned short __align__(32) pqkvb[3][16 * 192];
    __shared__ unsigned short __align__(32) qkvb[16 * 576];
    __shared__ float kx[16 * 3];
    __shared__ int   widx[16];
    __shared__ float sc16[16];
    __shared__ float attn[NHEAD * 16];

    int lane = threadIdx.x;
    int n = blockIdx.x;
    const unsigned short* wt = (const unsigned short*)(ws + OFF_WT);
    if (lane < 16) widx[lane] = knn[n * NSW + lane];
    __syncthreads();
    if (lane < 16) {
        int j = widx[lane];
        kx[lane * 3] = p[j * 3]; kx[lane * 3 + 1] = p[j * 3 + 1]; kx[lane * 3 + 2] = p[j * 3 + 2];
    }
    __syncthreads();
    // pairwise rel coords: pos[s][3j+c] = kxyz[j][c] - kxyz[s][c], padded 48->64
    for (int t = lane; t < 16 * 64; t += 32) {
        int s = t / 64, l = t % 64;
        float v = 0.f;
        if (l < 48) { int j = l / 3, c = l % 3; v = kx[j * 3 + c] - kx[s * 3 + c]; }
        posb[t] = f2bfu(v);
    }
    __syncthreads();
    const float* bns = &ws[OFF_BNS];
    const float* bnf = &ws[OFF_BNF];
    for (int i = 0; i < 3; i++) {
        for (int n0 = 0; n0 < 192; n0 += 16) {
            v8f cc = {};
            for (int k0 = 0; k0 < 64; k0 += 32) {
                v16bf a = load_a_frag(posb, 64, k0, lane);
                v16bf b = load_bt_frag(wt + WT_PW1 + i * 192 * 64, 64, k0, n0, lane);
                cc = wmma_bf16(a, b, cc);
            }
            int col = n0 + (lane & 15), rb = (lane >> 4) * 8;
            float b1 = pos_b1[i * 192 + col], s = bns[i * 192 + col], sh = bnf[i * 192 + col];
#pragma unroll
            for (int r = 0; r < 8; r++) {
                float h = (cc[r] + b1) * s + sh;
                h = h > 0.f ? h : 0.f;
                hnb[(rb + r) * 192 + col] = f2bfu(h);
            }
        }
        __syncthreads();
        for (int n0 = 0; n0 < 192; n0 += 16) {
            v8f cc = {};
            for (int k0 = 0; k0 < 192; k0 += 32) {
                v16bf a = load_a_frag(hnb, 192, k0, lane);
                v16bf b = load_bt_frag(wt + WT_PW2 + i * 192 * 192, 192, k0, n0, lane);
                cc = wmma_bf16(a, b, cc);
            }
            int col = n0 + (lane & 15), rb = (lane >> 4) * 8;
            float b2 = pos_b2[i * 192 + col];
#pragma unroll
            for (int r = 0; r < 8; r++) pqkvb[i][(rb + r) * 192 + col] = f2bfu(cc[r] + b2);
        }
        __syncthreads();
    }
    // gather fused features and compute qkv
    const float* xf = &ws[OFF_XF];
    for (int t = lane; t < 16 * 192; t += 32) {
        int r = t / 192, c = t % 192;
        wxb[t] = f2bfu(xf[widx[r] * DIMF + c]);
    }
    __syncthreads();
    for (int n0 = 0; n0 < 576; n0 += 16) {
        v8f cc = {};
        for (int k0 = 0; k0 < 192; k0 += 32) {
            v16bf a = load_a_frag(wxb, 192, k0, lane);
            v16bf b = load_bt_frag(wt + WT_QKV, 192, k0, n0, lane);
            cc = wmma_bf16(a, b, cc);
        }
        int col = n0 + (lane & 15), rb = (lane >> 4) * 8;
#pragma unroll
        for (int r = 0; r < 8; r++) qkvb[(rb + r) * 576 + col] = f2bfu(cc[r]);
    }
    __syncthreads();
    // attention (only window-center row i=0 is needed)
    const float scale = 0.1767766952966369f;  // 32^-0.5
    for (int h = 0; h < NHEAD; h++) {
        float sj = 0.f;
        if (lane < 16) {
            int j = lane;
            for (int d = 0; d < HD; d++) {
                int c = h * HD + d;
                float q0  = bfu2f(qkvb[c]);
                float k0v = bfu2f(qkvb[192 + c]);
                float kj  = bfu2f(qkvb[j * 576 + 192 + c]);
                float pqj = bfu2f(pqkvb[0][j * 192 + c]);
                float pkj = bfu2f(pqkvb[1][j * 192 + c]);
                sj += q0 * kj + q0 * pqj + k0v * pkj;
            }
            sj *= scale;
        }
        __syncthreads();
        if (lane < 16) sc16[lane] = sj;
        __syncthreads();
        float av = 0.f;
        if (lane < 16) {
            float mxv = -1e30f;
            for (int j = 0; j < 16; j++) mxv = fmaxf(mxv, sc16[j]);
            float den = 0.f;
            for (int j = 0; j < 16; j++) den += expf(sc16[j] - mxv);
            av = expf(sc16[lane] - mxv) / den;
        }
        __syncthreads();
        if (lane < 16) attn[h * 16 + lane] = av;
        __syncthreads();
    }
    float* ao = &ws[OFF_AO];
    for (int c = lane; c < 192; c += 32) {
        int h = c / HD;
        float o = 0.f;
        for (int j = 0; j < 16; j++)
            o += attn[h * 16 + j] * (bfu2f(qkvb[j * 576 + 384 + c]) + bfu2f(pqkvb[2][j * 192 + c]));
        ao[n * DIMF + c] = o;
    }
}

extern "C" void kernel_launch(void* const* d_in, const int* in_sizes, int n_in,
                              void* d_out, int out_size, void* d_ws, size_t ws_size,
                              hipStream_t stream) {
    const float* p      = (const float*)d_in[0];
    const float* x      = (const float*)d_in[1];
    const int*   knn    = (const int*)d_in[2];
    const float* W_qkv  = (const float*)d_in[3];
    const float* W_proj = (const float*)d_in[4];
    const float* b_proj = (const float*)d_in[5];
    const float* prop_W = (const float*)d_in[6];
    const float* prop_b = (const float*)d_in[7];
    const float* gA_W   = (const float*)d_in[8];
    const float* gA_b   = (const float*)d_in[9];
    const float* gB_W   = (const float*)d_in[10];
    const float* gB_b   = (const float*)d_in[11];
    const float* gC_W   = (const float*)d_in[12];
    const float* gC_b   = (const float*)d_in[13];
    const float* pos_W1 = (const float*)d_in[14];
    const float* pos_b1 = (const float*)d_in[15];
    const float* pos_g  = (const float*)d_in[16];
    const float* pos_be = (const float*)d_in[17];
    const float* pos_W2 = (const float*)d_in[18];
    const float* pos_b2 = (const float*)d_in[19];
    const float* kn     = (const float*)d_in[20];
    const float* de_W1  = (const float*)d_in[21];
    const float* de_b1  = (const float*)d_in[22];
    const float* de_W2  = (const float*)d_in[23];
    const float* de_b2  = (const float*)d_in[24];
    const float* kp_W   = (const float*)d_in[25];
    const float* kp_b   = (const float*)d_in[26];
    const float* fus_W  = (const float*)d_in[27];
    const float* fus_b  = (const float*)d_in[28];
    float* ws = (float*)d_ws;
    unsigned short* wt = (unsigned short*)(ws + OFF_WT);

    LGP_init<<<(OFF_PRE + 255) / 256, 256, 0, stream>>>(ws);
    // transpose+convert weights to bf16 [col][K] with K zero-padded
    LGP_prep<<<128, 256, 0, stream>>>(W_qkv,  wt + WT_QKV,  192, 576, 192, 1);
    LGP_prep<<<64, 256, 0, stream>>>(W_proj, wt + WT_PROJ, 192, 192, 192, 1);
    LGP_prep<<<64, 256, 0, stream>>>(prop_W, wt + WT_PROP, 195, 192, 224, 1);
    LGP_prep<<<16, 256, 0, stream>>>(de_W2,  wt + WT_DE2,   48, 192,  64, 1);
    LGP_prep<<<64, 256, 0, stream>>>(fus_W,  wt + WT_FUS,  192, 192, 192, 1);
    LGP_prep<<<64, 256, 0, stream>>>(pos_W1, wt + WT_PW1,   48, 192,  64, 3);
    LGP_prep<<<128, 256, 0, stream>>>(pos_W2, wt + WT_PW2,  192, 192, 192, 3);

    LGP_k1<<<N_PTS / 2, 64, 0, stream>>>(p, x, knn, prop_b, gA_W, gA_b, gB_W, gB_b,
                                         kn, de_W1, de_b1, de_b2, kp_W, kp_b, ws);
    LGP_k2<<<1, 256, 0, stream>>>(gC_W, gC_b, pos_W1, pos_b1, pos_g, pos_be, ws);
    LGP_gemm192<<<N_PTS / 128, 256, 0, stream>>>(ws + OFF_PRE, ws + OFF_VOL, wt + WT_FUS, fus_b,
                                                 ws + OFF_XF);
    LGP_k4<<<N_PTS, 32, 0, stream>>>(p, knn, pos_b1, pos_b2, ws);
    LGP_gemm192<<<N_PTS / 128, 256, 0, stream>>>(ws + OFF_AO, nullptr, wt + WT_PROJ, b_proj,
                                                 (float*)d_out);
}
